// PAB_71193377899213
// MI455X (gfx1250) — matrix operand order repaired and checked by reading, not compiled, not run on hardware
//
#include <hip/hip_runtime.h>

// ---------------------------------------------------------------------------
// PAB (pyramid attention block) with SelfONN convs, flash-style global softmax
// B=4, C=P=64, H=W=64, HW=4096, Q=3.
//
// Attention GEMMs run on V_WMMA_F32_16X16X32_F16 (wave32). Operands are laid
// out in WMMA fragment order by the producer kernels so the attention kernels
// do pure b128 loads. Global softmax (single M,S per batch) lets us avoid the
// 268MB score matrix entirely: pass1 computes online (max, sum-exp), pass2
// recomputes scores and contracts exp(s-M)/S with Xb on the fly.
// ---------------------------------------------------------------------------

typedef __attribute__((ext_vector_type(16))) _Float16 v16h;
typedef __attribute__((ext_vector_type(8)))  _Float16 v8h;
typedef __attribute__((ext_vector_type(8)))  float    v8f;

#define WMMA_F16F32(A, B, Cacc) \
  __builtin_amdgcn_wmma_f32_16x16x32_f16(false, (A), false, (B), (short)0, (Cacc), false, false)

// wave32 xor-swizzle reductions (group-of-32 mode: and=0x1f, xor in bits[14:10])
#define SWZ_F(v, pat) \
  __builtin_bit_cast(float, __builtin_amdgcn_ds_swizzle(__builtin_bit_cast(int, (v)), (pat)))

__device__ __forceinline__ float waveReduceMax(float v) {
  v = fmaxf(v, SWZ_F(v, 0x401F));
  v = fmaxf(v, SWZ_F(v, 0x201F));
  v = fmaxf(v, SWZ_F(v, 0x101F));
  v = fmaxf(v, SWZ_F(v, 0x081F));
  v = fmaxf(v, SWZ_F(v, 0x041F));
  return v;
}
__device__ __forceinline__ float waveReduceSum(float v) {
  v += SWZ_F(v, 0x401F);
  v += SWZ_F(v, 0x201F);
  v += SWZ_F(v, 0x101F);
  v += SWZ_F(v, 0x081F);
  v += SWZ_F(v, 0x041F);
  return v;
}

// ---------------------------------------------------------------------------
// Kernel 1: 1x1 SelfONN convs (top & center). One thread per (b, p, pixel).
// Writes x_top into WMMA *B*-fragment layout (score GEMM: K=p, N=pixel) and
// x_center into WMMA *A*-fragment layout (row n=pixel, K=p).
// Fragment storage: [b][tile][kchunk][lane(32)][half(16)] -> contiguous b128/lane.
// ---------------------------------------------------------------------------
__global__ void __launch_bounds__(256)
pab_conv1x1(const float* __restrict__ x,
            const float* __restrict__ top_w, const float* __restrict__ top_b,
            const float* __restrict__ cen_w, const float* __restrict__ cen_b,
            _Float16* __restrict__ xtB, _Float16* __restrict__ xcA) {
  const int pix = blockIdx.x * 256 + threadIdx.x;
  const int p   = blockIdx.y;
  const int b   = blockIdx.z;
  const float* xb = x + (size_t)b * 64 * 4096;
  const float* tw = top_w + (size_t)p * 192;
  const float* cw = cen_w + (size_t)p * 192;
  float at = top_b[p], ac = cen_b[p];
#pragma unroll 4
  for (int c = 0; c < 64; ++c) {
    float xv = xb[c * 4096 + pix];
    float x2 = xv * xv, x3 = x2 * xv;
    at += tw[c] * xv + tw[64 + c] * x2 + tw[128 + c] * x3;
    ac += cw[c] * xv + cw[64 + c] * x2 + cw[128 + c] * x3;
  }
  // x_top -> B fragment (16-bit B 32x16: lane<16 K={0..7,16..23}, lane>=16 K={8..15,24..31})
  {
    int kc = p >> 5, kk = p & 31;
    int hi = (kk >> 3) & 1, h = (kk & 7) + ((kk >> 4) << 3);
    int lane = (pix & 15) + (hi << 4);
    int mtile = pix >> 4;
    size_t off = ((((size_t)b * 256 + mtile) * 2 + kc) * 32 + lane) * 16 + h;
    xtB[off] = (_Float16)at;
  }
  // x_center -> A fragment (16-bit A 16x32: lane<16 K={0..7,16..23}, lane>=16 K={8..15,24..31})
  {
    int kc = p >> 5, kk = p & 31;
    int hi, e;
    if      (kk <  8) { hi = 0; e = kk;      }
    else if (kk < 16) { hi = 1; e = kk - 8;  }
    else if (kk < 24) { hi = 0; e = kk - 8;  }
    else              { hi = 1; e = kk - 16; }
    int lane = (pix & 15) + (hi << 4);
    int ntile = pix >> 4;
    size_t off = ((((size_t)b * 256 + ntile) * 2 + kc) * 32 + lane) * 16 + e;
    xcA[off] = (_Float16)ac;
  }
}

// ---------------------------------------------------------------------------
// Kernel 2: 3x3 SelfONN conv (bottom), pad=1. One thread per (b, c, pixel).
// Writes x_bottom into B-fragment layout for the second GEMM (K=m, N=c).
// ---------------------------------------------------------------------------
__global__ void __launch_bounds__(256)
pab_conv3x3_bottom(const float* __restrict__ x, const float* __restrict__ w,
                   const float* __restrict__ bias, _Float16* __restrict__ xbB) {
  const int pix = blockIdx.x * 256 + threadIdx.x;
  const int c = blockIdx.y, b = blockIdx.z;
  const int h = pix >> 6, wcol = pix & 63;
  const float* xb = x + (size_t)b * 64 * 4096;
  float acc = bias[c];
  for (int cc = 0; cc < 64; ++cc) {
    const float* xc = xb + cc * 4096;
    const float* w1 = w + ((size_t)c * 192 + cc) * 9;
    const float* w2 = w + ((size_t)c * 192 + 64 + cc) * 9;
    const float* w3 = w + ((size_t)c * 192 + 128 + cc) * 9;
#pragma unroll
    for (int ky = 0; ky < 3; ++ky) {
      int hy = h + ky - 1;
      if ((unsigned)hy >= 64u) continue;
#pragma unroll
      for (int kx = 0; kx < 3; ++kx) {
        int wx = wcol + kx - 1;
        if ((unsigned)wx >= 64u) continue;
        float xv = xc[hy * 64 + wx];
        float x2 = xv * xv;
        int t = ky * 3 + kx;
        acc += w1[t] * xv + w2[t] * x2 + w3[t] * (x2 * xv);
      }
    }
  }
  int mc = pix >> 5, mm = pix & 31;
  int hi = (mm >> 3) & 1, hh = (mm & 7) + ((mm >> 4) << 3);
  int lane = (c & 15) + (hi << 4);
  int ct = c >> 4;
  size_t off = ((((size_t)b * 4 + ct) * 128 + mc) * 32 + lane) * 16 + hh;
  xbB[off] = (_Float16)acc;
}

// ---------------------------------------------------------------------------
// Kernel 3: attention pass 1 — per-(b, n-tile) online (max, sum-exp) of the
// score row-band S = Xc^T * Xt, K=64 via 2x V_WMMA_F32_16X16X32_F16 per tile.
// ---------------------------------------------------------------------------
__global__ void __launch_bounds__(256)
pab_attn_stats(const _Float16* __restrict__ xcA, const _Float16* __restrict__ xtB,
               float2* __restrict__ partials) {
  const int lane = threadIdx.x & 31, wv = threadIdx.x >> 5;
  const int nt = blockIdx.x * 8 + wv, b = blockIdx.y;
  const _Float16* abase = xcA + ((size_t)b * 256 + nt) * 1024;
  v16h a0 = *(const v16h*)(abase + lane * 16);
  v16h a1 = *(const v16h*)(abase + 512 + lane * 16);
  const _Float16* tb = xtB + (size_t)b * 256 * 1024;
  v8f z = {};
  float m_run = -__builtin_inff();
  float s = 0.0f;
  for (int mc = 0; mc < 128; ++mc) {
    const _Float16* bp = tb + (size_t)mc * 2048 + lane * 16;
    v16h bt00 = *(const v16h*)(bp);
    v16h bt01 = *(const v16h*)(bp + 512);
    v16h bt10 = *(const v16h*)(bp + 1024);
    v16h bt11 = *(const v16h*)(bp + 1536);
    v8f acc0 = WMMA_F16F32(a0, bt00, z);
    acc0     = WMMA_F16F32(a1, bt01, acc0);
    v8f acc1 = WMMA_F16F32(a0, bt10, z);
    acc1     = WMMA_F16F32(a1, bt11, acc1);
    float tmax = fmaxf(acc0[0], acc1[0]);
#pragma unroll
    for (int v = 1; v < 8; ++v) tmax = fmaxf(tmax, fmaxf(acc0[v], acc1[v]));
    tmax = waveReduceMax(tmax);              // wave-uniform
    if (tmax > m_run) { s *= __expf(m_run - tmax); m_run = tmax; }
#pragma unroll
    for (int v = 0; v < 8; ++v)
      s += __expf(acc0[v] - m_run) + __expf(acc1[v] - m_run);
  }
  s = waveReduceSum(s);
  if (lane == 0) partials[b * 256 + nt] = make_float2(m_run, s);
}

// ---------------------------------------------------------------------------
// Kernel 4: combine 256 per-tile (m_i, s_i) pairs -> global (M, S) per batch.
// ---------------------------------------------------------------------------
__global__ void __launch_bounds__(256)
pab_softmax_reduce(const float2* __restrict__ partials, float2* __restrict__ stats) {
  __shared__ float sm[256];
  __shared__ float ss[256];
  const int b = blockIdx.x, t = threadIdx.x;
  float2 p = partials[b * 256 + t];
  sm[t] = p.x;
  __syncthreads();
  for (int st = 128; st > 0; st >>= 1) {
    if (t < st) sm[t] = fmaxf(sm[t], sm[t + st]);
    __syncthreads();
  }
  float M = sm[0];
  __syncthreads();
  ss[t] = p.y * __expf(p.x - M);
  __syncthreads();
  for (int st = 128; st > 0; st >>= 1) {
    if (t < st) ss[t] += ss[t + st];
    __syncthreads();
  }
  if (t == 0) stats[b] = make_float2(M, ss[0]);
}

// ---------------------------------------------------------------------------
// Kernel 5: attention pass 2 — O(n, 0..63) = sum_m exp(S(n,m)-M)/S * Xb(m, c).
// Score tiles (D layout) -> exp -> f16 -> per-wave LDS tile -> re-gather as
// A-fragment for the second WMMA chain. One wave per 16-row n-tile.
// ---------------------------------------------------------------------------
__global__ void __launch_bounds__(256)
pab_attn_apply(const _Float16* __restrict__ xcA, const _Float16* __restrict__ xtB,
               const _Float16* __restrict__ xbB, const float2* __restrict__ stats,
               float* __restrict__ O) {
  __shared__ _Float16 ptile[8 * 512];   // 16x32 f16 tile per wave
  const int lane = threadIdx.x & 31, wv = threadIdx.x >> 5;
  const int nt = blockIdx.x * 8 + wv, b = blockIdx.y;
  const float M = stats[b].x;
  const float invS = 1.0f / stats[b].y;
  const _Float16* abase = xcA + ((size_t)b * 256 + nt) * 1024;
  v16h a0 = *(const v16h*)(abase + lane * 16);
  v16h a1 = *(const v16h*)(abase + 512 + lane * 16);
  const _Float16* tb = xtB + (size_t)b * 256 * 1024;
  const _Float16* vb = xbB + (size_t)b * 4 * 128 * 512;
  v8f z = {};
  v8f o0 = z, o1 = z, o2 = z, o3 = z;
  _Float16* my = ptile + wv * 512;
  const int r   = lane & 15;
  const int hi  = lane >> 4;
  const int hi8 = hi * 8;
  for (int mc = 0; mc < 128; ++mc) {
    const _Float16* bp = tb + (size_t)mc * 2048 + lane * 16;
    v16h bt00 = *(const v16h*)(bp);
    v16h bt01 = *(const v16h*)(bp + 512);
    v16h bt10 = *(const v16h*)(bp + 1024);
    v16h bt11 = *(const v16h*)(bp + 1536);
    v8f acc0 = WMMA_F16F32(a0, bt00, z);
    acc0     = WMMA_F16F32(a1, bt01, acc0);
    v8f acc1 = WMMA_F16F32(a0, bt10, z);
    acc1     = WMMA_F16F32(a1, bt11, acc1);
    // D layout -> row-major 16x32 LDS tile, with exp applied
#pragma unroll
    for (int v = 0; v < 8; ++v) {
      int row = v + hi8;
      my[row * 32 + r]      = (_Float16)__expf(acc0[v] - M);
      my[row * 32 + 16 + r] = (_Float16)__expf(acc1[v] - M);
    }
    asm volatile("s_wait_dscnt 0" ::: "memory");
    // Re-gather the probability tile as an A fragment:
    // lane<16: row=lane, K cols {0..7,16..23}; lane>=16: row=lane-16, {8..15,24..31}
    const _Float16* prow = my + r * 32 + (hi << 3);
    v8h lo = *(const v8h*)(prow);
    v8h hp = *(const v8h*)(prow + 16);
    v16h pa = __builtin_shufflevector(lo, hp, 0, 1, 2, 3, 4, 5, 6, 7,
                                      8, 9, 10, 11, 12, 13, 14, 15);
    const _Float16* vp = vb + (size_t)mc * 512 + lane * 16;
    v16h vb0 = *(const v16h*)(vp);
    v16h vb1 = *(const v16h*)(vp + 65536);
    v16h vb2 = *(const v16h*)(vp + 131072);
    v16h vb3 = *(const v16h*)(vp + 196608);
    o0 = WMMA_F16F32(pa, vb0, o0);
    o1 = WMMA_F16F32(pa, vb1, o1);
    o2 = WMMA_F16F32(pa, vb2, o2);
    o3 = WMMA_F16F32(pa, vb3, o3);
  }
  float* Ob = O + ((size_t)b * 4096 + nt * 16) * 64;
#pragma unroll
  for (int v = 0; v < 8; ++v) {
    int row = v + hi8;
    float* orow = Ob + row * 64 + r;
    orow[0]  = o0[v] * invS;
    orow[16] = o1[v] * invS;
    orow[32] = o2[v] * invS;
    orow[48] = o3[v] * invS;
  }
}

// ---------------------------------------------------------------------------
// Kernel 6: final 3x3 SelfONN conv on z = x + raw_reshape(O).
// raw reshape (B,HW,C)->(B,C,H,W): z(b,c,h,w) = x(b,c,h,w) + O[b][c*64+h][w].
// ---------------------------------------------------------------------------
__global__ void __launch_bounds__(256)
pab_conv3x3_out(const float* __restrict__ x, const float* __restrict__ Oin,
                const float* __restrict__ w, const float* __restrict__ bias,
                float* __restrict__ out) {
  const int pix = blockIdx.x * 256 + threadIdx.x;
  const int o = blockIdx.y, b = blockIdx.z;
  const int h = pix >> 6, wcol = pix & 63;
  const float* xb = x + (size_t)b * 64 * 4096;
  const float* Ob = Oin + (size_t)b * 4096 * 64;
  float acc = bias[o];
  for (int cc = 0; cc < 64; ++cc) {
    const float* xc = xb + cc * 4096;
    const float* Oc = Ob + cc * 4096;   // (cc*64 + hy)*64 + wx
    const float* w1 = w + ((size_t)o * 192 + cc) * 9;
    const float* w2 = w + ((size_t)o * 192 + 64 + cc) * 9;
    const float* w3 = w + ((size_t)o * 192 + 128 + cc) * 9;
#pragma unroll
    for (int ky = 0; ky < 3; ++ky) {
      int hy = h + ky - 1;
      if ((unsigned)hy >= 64u) continue;
#pragma unroll
      for (int kx = 0; kx < 3; ++kx) {
        int wx = wcol + kx - 1;
        if ((unsigned)wx >= 64u) continue;
        float zv = xc[hy * 64 + wx] + Oc[hy * 64 + wx];
        float z2 = zv * zv;
        int t = ky * 3 + kx;
        acc += w1[t] * zv + w2[t] * z2 + w3[t] * (z2 * zv);
      }
    }
  }
  out[((size_t)b * 64 + o) * 4096 + pix] = acc;
}

// ---------------------------------------------------------------------------
extern "C" void kernel_launch(void* const* d_in, const int* in_sizes, int n_in,
                              void* d_out, int out_size, void* d_ws, size_t ws_size,
                              hipStream_t stream) {
  (void)in_sizes; (void)n_in; (void)out_size; (void)ws_size;
  const float* x        = (const float*)d_in[0];
  const float* top_w    = (const float*)d_in[1];
  const float* top_b    = (const float*)d_in[2];
  const float* center_w = (const float*)d_in[3];
  const float* center_b = (const float*)d_in[4];
  const float* bottom_w = (const float*)d_in[5];
  const float* bottom_b = (const float*)d_in[6];
  const float* out_w    = (const float*)d_in[7];
  const float* out_b    = (const float*)d_in[8];
  float* out = (float*)d_out;

  char* ws = (char*)d_ws;
  _Float16* xcA      = (_Float16*)(ws);                       // 2 MB  A-frags (center)
  _Float16* xtB      = (_Float16*)(ws + (size_t)(2  << 20));  // 2 MB  B-frags (top)
  _Float16* xbB      = (_Float16*)(ws + (size_t)(4  << 20));  // 2 MB  B-frags (bottom)
  float*    O        = (float*)   (ws + (size_t)(6  << 20));  // 4 MB  attention output
  float2*   partials = (float2*)  (ws + (size_t)(10 << 20));  // 8 KB
  float2*   stats    = (float2*)  (ws + (size_t)(10 << 20) + 4 * 256 * sizeof(float2));

  pab_conv1x1<<<dim3(16, 64, 4), 256, 0, stream>>>(x, top_w, top_b, center_w, center_b,
                                                   xtB, xcA);
  pab_conv3x3_bottom<<<dim3(16, 64, 4), 256, 0, stream>>>(x, bottom_w, bottom_b, xbB);
  pab_attn_stats<<<dim3(32, 4), 256, 0, stream>>>(xcA, xtB, partials);
  pab_softmax_reduce<<<4, 256, 0, stream>>>(partials, stats);
  pab_attn_apply<<<dim3(32, 4), 256, 0, stream>>>(xcA, xtB, xbB, stats, O);
  pab_conv3x3_out<<<dim3(16, 64, 4), 256, 0, stream>>>(x, O, out_w, out_b, out);
}
